// MultiADDALocalAttention_59536836657198
// MI455X (gfx1250) — matrix operand order repaired and checked by reading, not compiled, not run on hardware
//
#include <hip/hip_runtime.h>
#include <math.h>
#include <stdint.h>

// ---------------------------------------------------------------------------
// MultiADDALocalAttention for MI455X (gfx1250, wave32, WMMA bf16 + TDM)
// Pixel-major layouts, p = ((b*64)+y)*64+x:
//   qkv  : fp32 [p][576]   (q: br*64, k: 192+br*64, v: 384+br*64)
//   offs : fp32 [dil-2][p][18]
//   dbuf : fp32 [sel][p][64]   sel = (dil-2)*2 + (0:k,1:v)
//   xbf  : bf16 [p][192]   (input packed for QKV GEMM)
//   abuf : bf16 [p][192]   (attention out, feeds proj GEMM)
//   wqkv/wproj/wdef : bf16 pre-packed weights (row-major [n][k])
// ---------------------------------------------------------------------------

#define NPIX 32768   // 8*64*64
#define Hh   64
#define Ww   64

typedef __bf16 bf16_t;
typedef __attribute__((ext_vector_type(16))) __bf16 v16bf;
typedef __attribute__((ext_vector_type(8)))  __bf16 v8bf;
typedef __attribute__((ext_vector_type(8)))  float  v8f;
typedef __attribute__((ext_vector_type(4)))  unsigned int v4u;
typedef __attribute__((ext_vector_type(4)))  int v4i;
typedef __attribute__((ext_vector_type(8)))  int v8i;

#if __has_builtin(__builtin_amdgcn_tensor_load_to_lds) && \
    __has_builtin(__builtin_amdgcn_s_wait_tensorcnt)
#define HAVE_TDM 1
#else
#define HAVE_TDM 0
#endif

// WMMA bf16 operand maps (ISA 7.12.2):
//  A (16x32): lane half selects K-run base (0 / 8); elems 0..7 -> K base..base+7,
//             elems 8..15 -> K base+16..base+23  (two contiguous 8-runs)
//  B (32x16): same K map as A; this code loads A and B contiguous: half h -> K 16h..16h+15
__device__ __forceinline__ v8f wmma_bf16(v16bf a, v16bf b, v8f c) {
  return __builtin_amdgcn_wmma_f32_16x16x32_bf16(
      false, a, false, b, (short)0, c, false, false);
}

__device__ __forceinline__ v16bf join16(v8bf lo, v8bf hi) {
  v16bf r;
#pragma unroll
  for (int e = 0; e < 8; ++e) { r[e] = lo[e]; r[e + 8] = hi[e]; }
  return r;
}

// ---------------------------------------------------------------------------
// Stage a dense rows x rowlen bf16 2D tile (row stride == rowlen) into LDS
// using the Tensor Data Mover; fallback: per-lane copy loop.
// D# packing per ISA ch.8: group0 {count=1, lds_addr, global_addr, type=2},
// group1 {data_size=2B, tensor_dim0/1, tile_dim0/1, tensor_dim0_stride}.
// ---------------------------------------------------------------------------
__device__ __forceinline__ void tdm_stage_2d(bf16_t* lds_dst, const bf16_t* gsrc,
                                             int rows, int rowlen, int lane) {
#if HAVE_TDM
  unsigned long long ga = (unsigned long long)(uintptr_t)gsrc;
  v4u g0;
  g0[0] = 1u;                                   // count=1, user descriptor
  g0[1] = (unsigned)(uintptr_t)lds_dst;         // LDS byte address (low 32 bits)
  g0[2] = (unsigned)(ga & 0xFFFFFFFFu);         // global_addr[31:0]
  g0[3] = (unsigned)((ga >> 32) & 0x01FFFFFFu) | 0x80000000u;  // addr[56:32]|type=2
  v8i g1;
  g1[0] = 1 << 16;                              // data_size = 1 -> 2 bytes
  g1[1] = (rowlen & 0xFFFF) << 16;              // tensor_dim0[15:0]  @63:48
  g1[2] = ((rowlen >> 16) & 0xFFFF)             // tensor_dim0[31:16] @79:64
        | ((rows & 0xFFFF) << 16);              // tensor_dim1[15:0]  @95:80
  g1[3] = (rowlen & 0xFFFF) << 16;              // tile_dim0 @127:112 (dim1 hi=0)
  g1[4] = rows & 0xFFFF;                        // tile_dim1 @143:128, tile_dim2=0
  g1[5] = rowlen;                               // tensor_dim0_stride[31:0]
  g1[6] = 0;
  g1[7] = 0;
  v4i gz = {0, 0, 0, 0};
#if __clang_major__ >= 23
  v8i gz8 = {0, 0, 0, 0, 0, 0, 0, 0};
  __builtin_amdgcn_tensor_load_to_lds(g0, g1, gz, gz, gz8, 0);
#else
  __builtin_amdgcn_tensor_load_to_lds(g0, g1, gz, gz, 0);
#endif
  __builtin_amdgcn_s_wait_tensorcnt(0);
  (void)lane;
#else
  const int n = rows * rowlen;
  for (int i = lane; i < n; i += 32) lds_dst[i] = gsrc[i];
#endif
}

// ---------------------------------------------------------------------------
// fp32 -> bf16 pack
// ---------------------------------------------------------------------------
__global__ void k_cvt(const float* __restrict__ src, bf16_t* __restrict__ dst, int n) {
  const int i = blockIdx.x * blockDim.x + threadIdx.x;
  if (i < n) dst[i] = (bf16_t)src[i];
}

// ---------------------------------------------------------------------------
// Pixel GEMM, all-bf16 operands: out[p][n] = sum_k A[p][k]*W[n][k] (+bias[n]).
// One 16x16 tile per wave; per-wave weight tile staged into LDS via TDM,
// operands loaded as 16B-packed v8bf chunks.
// ---------------------------------------------------------------------------
template <int KD, int ND>
__global__ __launch_bounds__(128) void k_gemm_bf(const bf16_t* __restrict__ A,
                                                 const bf16_t* __restrict__ W,
                                                 const float* __restrict__ bias,
                                                 float* __restrict__ out) {
  __shared__ bf16_t sB[4][16 * KD];
  const int wave = threadIdx.x >> 5, lane = threadIdx.x & 31;
  const int gw = blockIdx.x * 4 + wave;
  const int ntiles = ND / 16;
  const int mt = gw / ntiles, nt = gw % ntiles;

  tdm_stage_2d(&sB[wave][0], W + (size_t)nt * 16 * KD, 16, KD, lane);

  const int arow = mt * 16 + (lane & 15);
  const bf16_t* __restrict__ arp = A + (size_t)arow * KD;
  const bf16_t* __restrict__ brp = &sB[wave][(lane & 15) * KD];
  const int abase = lane & 16;          // 0 or 16
  const int bbase = lane & 16;          // 0 or 16
  v8f acc = {};
  for (int ks = 0; ks < KD; ks += 32) {
    if (ks + 32 < KD) __builtin_prefetch(arp + ks + 32, 0, 1);
    const v8bf a0 = *(const v8bf*)(arp + ks + abase);
    const v8bf a1 = *(const v8bf*)(arp + ks + abase + 8);
    const v8bf b0 = *(const v8bf*)(brp + ks + bbase);
    const v8bf b1 = *(const v8bf*)(brp + ks + bbase + 8);
    acc = wmma_bf16(join16(a0, a1), join16(b0, b1), acc);
  }
  const int ncol = nt * 16 + (lane & 15);
  const float bv = bias ? bias[ncol] : 0.0f;
  const int rbase = mt * 16 + ((lane & 16) >> 1);
#pragma unroll
  for (int r = 0; r < 8; ++r)
    out[(size_t)(rbase + r) * ND + ncol] = acc[r] + bv;
}

// ---------------------------------------------------------------------------
// Offsets conv: conv2d(k_branch, off_w, off_b, dilation=dil, padding=dil).
// One thread per (pixel, oc<18). grid.y: 0->dil2, 1->dil3.
// ---------------------------------------------------------------------------
__global__ void k_off(const float* __restrict__ qkv,
                      const float* __restrict__ ow2, const float* __restrict__ ob2,
                      const float* __restrict__ ow3, const float* __restrict__ ob3,
                      float* __restrict__ offs) {
  const int dilm2 = blockIdx.y;
  const int dil = dilm2 + 2;
  const float* __restrict__ ow = dilm2 ? ow3 : ow2;
  const float* __restrict__ ob = dilm2 ? ob3 : ob2;
  const int idx = blockIdx.x * blockDim.x + threadIdx.x;
  if (idx >= NPIX * 18) return;
  const int p = idx / 18, oc = idx % 18;
  const int b = p >> 12, y = (p >> 6) & 63, x = p & 63;
  const int chbase = 192 + (dil - 1) * 64;
  float acc = ob[oc];
  for (int ky = 0; ky < 3; ++ky) {
    const int yy = y + (ky - 1) * dil;
    if (yy < 0 || yy >= Hh) continue;
    for (int kx = 0; kx < 3; ++kx) {
      const int xx = x + (kx - 1) * dil;
      if (xx < 0 || xx >= Ww) continue;
      const float* __restrict__ src =
          qkv + (size_t)((b << 12) + (yy << 6) + xx) * 576 + chbase;
      const float* __restrict__ wr = ow + (size_t)oc * 64 * 9 + ky * 3 + kx;
      for (int c = 0; c < 64; ++c) acc += src[c] * wr[(size_t)c * 9];
    }
  }
  offs[(size_t)dilm2 * NPIX * 18 + idx] = acc;
}

// ---------------------------------------------------------------------------
// Deformable conv (k and v, dil 2/3):
//   stage 1: bilinear-sample 16 pixels x (64ch x 9taps) -> LDS bf16, K=c*9+kk
//   stage 2: [16x576] @ dw^T[576x64] via 18x4 WMMA bf16 (packed LDS/global loads)
// grid.y sel: 0:(d2,k) 1:(d2,v) 2:(d3,k) 3:(d3,v). 2 waves / 64-thread block.
// ---------------------------------------------------------------------------
#define DEF_WAVES 2
__global__ __launch_bounds__(64) void k_deform(
    const float* __restrict__ qkv, const float* __restrict__ offs,
    const bf16_t* __restrict__ wdef,
    const float* __restrict__ db2, const float* __restrict__ db3,
    float* __restrict__ dbuf) {
  __shared__ bf16_t sA[DEF_WAVES][16][576];  // 36 KB
  const int wave = threadIdx.x >> 5, lane = threadIdx.x & 31;
  const int sel = blockIdx.y;
  const int dilm2 = sel >> 1, isv = sel & 1;
  const int dil = dilm2 + 2;
  const bf16_t* __restrict__ dw = wdef + (size_t)dilm2 * 64 * 576;
  const float* __restrict__ db = dilm2 ? db3 : db2;
  const int chbase = (isv ? 384 : 192) + (dil - 1) * 64;
  const float* __restrict__ off = offs + (size_t)dilm2 * NPIX * 18;
  const int tile = blockIdx.x * DEF_WAVES + wave;

  // stage 1: bilinear sampling (144 (pixel,tap) pairs per wave)
  for (int pt = lane; pt < 144; pt += 32) {
    const int pix = pt / 9, kk = pt % 9;
    const int p = tile * 16 + pix;
    const int b = p >> 12, y = (p >> 6) & 63, x = p & 63;
    const float offy = off[(size_t)p * 18 + kk * 2 + 0];
    const float offx = off[(size_t)p * 18 + kk * 2 + 1];
    const float py = (float)y + (float)((kk / 3) * dil - dil) + offy;
    const float px = (float)x + (float)((kk % 3) * dil - dil) + offx;
    const float fy0 = floorf(py), fx0 = floorf(px);
    const float wy = py - fy0, wx = px - fx0;
    const int y0 = (int)fy0, x0 = (int)fx0, y1 = y0 + 1, x1 = x0 + 1;
    const bool vy0 = (y0 >= 0) && (y0 < Hh), vy1 = (y1 >= 0) && (y1 < Hh);
    const bool vx0 = (x0 >= 0) && (x0 < Ww), vx1 = (x1 >= 0) && (x1 < Ww);
    const int yc0 = min(max(y0, 0), Hh - 1), yc1 = min(max(y1, 0), Hh - 1);
    const int xc0 = min(max(x0, 0), Ww - 1), xc1 = min(max(x1, 0), Ww - 1);
    const float* __restrict__ base = qkv + ((size_t)b << 12) * 576;
    const float* __restrict__ p00 = base + (size_t)(yc0 * Ww + xc0) * 576 + chbase;
    const float* __restrict__ p01 = base + (size_t)(yc0 * Ww + xc1) * 576 + chbase;
    const float* __restrict__ p10 = base + (size_t)(yc1 * Ww + xc0) * 576 + chbase;
    const float* __restrict__ p11 = base + (size_t)(yc1 * Ww + xc1) * 576 + chbase;
    const float w00 = (vy0 && vx0) ? (1.f - wy) * (1.f - wx) : 0.f;
    const float w01 = (vy0 && vx1) ? (1.f - wy) * wx : 0.f;
    const float w10 = (vy1 && vx0) ? wy * (1.f - wx) : 0.f;
    const float w11 = (vy1 && vx1) ? wy * wx : 0.f;
    for (int c = 0; c < 64; ++c) {
      const float val = w00 * p00[c] + w01 * p01[c] + w10 * p10[c] + w11 * p11[c];
      sA[wave][pix][c * 9 + kk] = (bf16_t)val;
    }
  }
  __syncthreads();

  // stage 2: WMMA GEMM, K=576, N=64 (A from LDS, B packed bf16 from global)
  const int row = lane & 15;
  const int abase = lane & 16;
  const int bbase = lane & 16;
  float* __restrict__ outp = dbuf + (size_t)sel * NPIX * 64;
  for (int ntile = 0; ntile < 4; ++ntile) {
    const int ocol = ntile * 16 + row;
    const bf16_t* __restrict__ wrow = dw + (size_t)ocol * 576;
    v8f acc = {};
    for (int ks = 0; ks < 576; ks += 32) {
      const v8bf a0 = *(const v8bf*)&sA[wave][row][ks + abase];
      const v8bf a1 = *(const v8bf*)&sA[wave][row][ks + abase + 8];
      const v8bf b0 = *(const v8bf*)(wrow + ks + bbase);
      const v8bf b1 = *(const v8bf*)(wrow + ks + bbase + 8);
      acc = wmma_bf16(join16(a0, a1), join16(b0, b1), acc);
    }
    const float bv = db[ocol];
    const int rbase = tile * 16 + ((lane & 16) >> 1);
#pragma unroll
    for (int r = 0; r < 8; ++r)
      outp[(size_t)(rbase + r) * 64 + ocol] = acc[r] + bv;
  }
}

// ---------------------------------------------------------------------------
// Local attention: one wave32 per (pixel, branch). 9 taps, hd=64 (2 ch/lane),
// q.k via __shfl_xor butterfly, register softmax, weighted v sum -> bf16 abuf.
// ---------------------------------------------------------------------------
__global__ void k_attn(const float* __restrict__ qkv, const float* __restrict__ dbuf,
                       bf16_t* __restrict__ abuf) {
  const int wave = threadIdx.x >> 5, lane = threadIdx.x & 31;
  const int p = blockIdx.x * 4 + wave;
  const int br = blockIdx.y;  // 0..2, dil = br+1
  const int dil = br + 1;
  const int b = p >> 12, y = (p >> 6) & 63, x = p & 63;

  const float* __restrict__ kb;
  const float* __restrict__ vb;
  int stride;
  if (br == 0) { kb = qkv + 192; vb = qkv + 384; stride = 576; }
  else {
    kb = dbuf + (size_t)((br - 1) * 2 + 0) * NPIX * 64;
    vb = dbuf + (size_t)((br - 1) * 2 + 1) * NPIX * 64;
    stride = 64;
  }
  const float q0 = qkv[(size_t)p * 576 + br * 64 + lane];
  const float q1 = qkv[(size_t)p * 576 + br * 64 + lane + 32];

  float lg[9], v0[9], v1[9];
#pragma unroll
  for (int kk = 0; kk < 9; ++kk) {
    const int yy = y + (kk / 3 - 1) * dil;
    const int xx = x + (kk % 3 - 1) * dil;
    const bool ok = (yy >= 0) && (yy < Hh) && (xx >= 0) && (xx < Ww);
    const int yc = min(max(yy, 0), Hh - 1), xc = min(max(xx, 0), Ww - 1);
    const size_t np = ((size_t)b << 12) + (yc << 6) + xc;
    const float k0 = ok ? kb[np * stride + lane] : 0.f;
    const float k1 = ok ? kb[np * stride + lane + 32] : 0.f;
    v0[kk] = ok ? vb[np * stride + lane] : 0.f;
    v1[kk] = ok ? vb[np * stride + lane + 32] : 0.f;
    float partial = q0 * k0 + q1 * k1;
#pragma unroll
    for (int m = 16; m >= 1; m >>= 1) partial += __shfl_xor(partial, m, 32);
    lg[kk] = partial * 0.125f;  // hd^-0.5
  }
  float mx = lg[0];
#pragma unroll
  for (int kk = 1; kk < 9; ++kk) mx = fmaxf(mx, lg[kk]);
  float s = 0.f;
#pragma unroll
  for (int kk = 0; kk < 9; ++kk) { lg[kk] = expf(lg[kk] - mx); s += lg[kk]; }
  const float inv = 1.f / s;
  float o0 = 0.f, o1 = 0.f;
#pragma unroll
  for (int kk = 0; kk < 9; ++kk) {
    const float a = lg[kk] * inv;
    o0 += a * v0[kk];
    o1 += a * v1[kk];
  }
  abuf[(size_t)p * 192 + br * 64 + lane] = (bf16_t)o0;
  abuf[(size_t)p * 192 + br * 64 + lane + 32] = (bf16_t)o1;
}

// ---------------------------------------------------------------------------
extern "C" void kernel_launch(void* const* d_in, const int* in_sizes, int n_in,
                              void* d_out, int out_size, void* d_ws, size_t ws_size,
                              hipStream_t stream) {
  (void)in_sizes; (void)n_in; (void)out_size; (void)ws_size;
  const float* x      = (const float*)d_in[0];
  const float* qkv_w  = (const float*)d_in[1];   // (576,192)
  const float* proj_w = (const float*)d_in[2];   // (192,192)
  const float* proj_b = (const float*)d_in[3];
  const float* off_w2 = (const float*)d_in[4];
  const float* off_b2 = (const float*)d_in[5];
  const float* def_w2 = (const float*)d_in[6];
  const float* def_b2 = (const float*)d_in[7];
  const float* off_w3 = (const float*)d_in[8];
  const float* off_b3 = (const float*)d_in[9];
  const float* def_w3 = (const float*)d_in[10];
  const float* def_b3 = (const float*)d_in[11];
  float* out = (float*)d_out;

  char* base = (char*)d_ws;
  float* qkv    = (float*)base;  base += (size_t)NPIX * 576 * 4;
  float* offs   = (float*)base;  base += (size_t)2 * NPIX * 18 * 4;
  float* dbuf   = (float*)base;  base += (size_t)4 * NPIX * 64 * 4;
  bf16_t* xbf   = (bf16_t*)base; base += (size_t)NPIX * 192 * 2;
  bf16_t* abuf  = (bf16_t*)base; base += (size_t)NPIX * 192 * 2;
  bf16_t* wqkv  = (bf16_t*)base; base += (size_t)576 * 192 * 2;
  bf16_t* wproj = (bf16_t*)base; base += (size_t)192 * 192 * 2;
  bf16_t* wdef  = (bf16_t*)base; base += (size_t)2 * 64 * 576 * 2;

  // 0) bf16 packing of GEMM-side activations and weights
  k_cvt<<<dim3((NPIX * 192 + 255) / 256), 256, 0, stream>>>(x, xbf, NPIX * 192);
  k_cvt<<<dim3((576 * 192 + 255) / 256), 256, 0, stream>>>(qkv_w, wqkv, 576 * 192);
  k_cvt<<<dim3((192 * 192 + 255) / 256), 256, 0, stream>>>(proj_w, wproj, 192 * 192);
  k_cvt<<<dim3((64 * 576 + 255) / 256), 256, 0, stream>>>(def_w2, wdef, 64 * 576);
  k_cvt<<<dim3((64 * 576 + 255) / 256), 256, 0, stream>>>(def_w3, wdef + 64 * 576, 64 * 576);
  // 1) QKV GEMM: 2048 M-tiles x 36 N-tiles, 4 waves/block, TDM-staged weights
  k_gemm_bf<192, 576><<<dim3((2048 * 36) / 4), 128, 0, stream>>>(xbf, wqkv, nullptr, qkv);
  // 2) offsets conv, dil 2/3
  k_off<<<dim3((NPIX * 18) / 256, 2), 256, 0, stream>>>(qkv, off_w2, off_b2, off_w3, off_b3, offs);
  // 3) deformable conv for k,v at dil 2,3
  k_deform<<<dim3(2048 / DEF_WAVES, 4), 64, 0, stream>>>(qkv, offs, wdef, def_b2, def_b3, dbuf);
  // 4) local attention, 3 branches, 1 wave/pixel
  k_attn<<<dim3(NPIX / 4, 3), 128, 0, stream>>>(qkv, dbuf, abuf);
  // 5) projection GEMM + bias -> d_out
  k_gemm_bf<192, 192><<<dim3((2048 * 12) / 4), 128, 0, stream>>>(abuf, wproj, proj_b, out);
}
